// Block_21242908246357
// MI455X (gfx1250) — matrix operand order repaired and checked by reading, not compiled, not run on hardware
//
#include <hip/hip_runtime.h>

// ---------------- constants ----------------
#define NE   512          // embed dim
#define NH   8            // heads
#define HD   64           // head dim
#define BB   64           // batch
#define TT   312          // seq len (q and kv)
#define TP   320          // seq len padded to 16
#define MTOK (BB * TT)    // 19968 rows, divisible by 128
#define NEGV (-1e30f)

typedef unsigned short u16;
typedef unsigned long long u64;
typedef __attribute__((ext_vector_type(16))) __bf16 v16bf;
typedef __attribute__((ext_vector_type(8)))  float  v8f;
typedef int v4i __attribute__((vector_size(16)));   // matches builtin param type

#if __has_builtin(__builtin_amdgcn_global_load_async_to_lds_b128)
#define HAVE_ASYNC_LDS 1
#else
#define HAVE_ASYNC_LDS 0
#endif
#define AS1 __attribute__((address_space(1)))
#define AS3 __attribute__((address_space(3)))

// ---------------- helpers ----------------
__device__ __forceinline__ u16 f2bf(float f) {
  unsigned u = __builtin_bit_cast(unsigned, f);
  u += 0x7fffu + ((u >> 16) & 1u);            // round-to-nearest-even
  return (u16)(u >> 16);
}

// Build a 16 x bf16 fragment from two contiguous 16-byte runs.
__device__ __forceinline__ v16bf frag16(const u16* p0, const u16* p1) {
  union { uint4 u[2]; v16bf v; } w;
  w.u[0] = *(const uint4*)p0;
  w.u[1] = *(const uint4*)p1;
  return w.v;
}

// 4x4 bf16 register transpose: rows r[0..3] (4 elems each) -> cols out via shifts
__device__ __forceinline__ u64 tr4(u64 r0, u64 r1, u64 r2, u64 r3, int c) {
  return  ((r0 >> (16 * c)) & 0xffffull)
       | (((r1 >> (16 * c)) & 0xffffull) << 16)
       | (((r2 >> (16 * c)) & 0xffffull) << 32)
       | (((r3 >> (16 * c)) & 0xffffull) << 48);
}

// ---------------- fp32 -> bf16 cast ----------------
__global__ void cast_f32_bf16(const float* __restrict__ src, u16* __restrict__ dst, int n) {
  int i = blockIdx.x * blockDim.x + threadIdx.x;
  int stride = gridDim.x * blockDim.x;
  for (; i < n; i += stride) dst[i] = f2bf(src[i]);
}

// ---------------- WMMA bf16 GEMM ----------------
// C[M,N] = A[M,K] * B[K,N] (+bias) (ReLU opt). A,B bf16 row-major.
// Block tile: 128(M) x 64(N), K-step 32. 4 waves, each wave does 32x64 (8 WMMA/K-step).
// Output row remap (per-batch padded buffers): orow = (m/Tsrc)*Tdst + m%Tsrc when Tdst>0.
__global__ __launch_bounds__(128)
void gemm_bf16(const u16* __restrict__ A, const u16* __restrict__ Bw,
               const float* __restrict__ bias,
               float* __restrict__ outF, u16* __restrict__ outB,
               int M, int N, int K, int relu, int Tsrc, int Tdst)
{
  __shared__ __align__(16) u16 As[128][48];  // 128 rows x 32 K (pad->48, 96B row)
  __shared__ __align__(16) u16 Bs[64][48];   // transposed: 64 cols x 32 K

  const int tid  = threadIdx.x;
  const int lane = tid & 31;
  const int wave = tid >> 5;
  const int hf   = lane >> 4;                // K-half selector per ISA layout
  const int ln   = lane & 15;                // row (A) / col (B) within tile
  const long mbase = (long)blockIdx.x * 128;
  const long nbase = (long)blockIdx.y * 64;

  v8f acc[2][4];
  #pragma unroll
  for (int s = 0; s < 2; ++s)
    #pragma unroll
    for (int g = 0; g < 4; ++g) acc[s][g] = v8f{};

  // A staging: thread t stages global row (mbase+t), 32 shorts per K-step
  const u16* ag = A + (mbase + tid) * (long)K;
  // B staging: thread t handles 4x4 block (kb, cb)
  const int kb = (tid & 7) * 4;
  const int cb = (tid >> 3) * 4;

  for (int k0 = 0; k0 < K; k0 += 32) {
    // ---- stage A (straight copy: async-to-LDS when available) ----
#if HAVE_ASYNC_LDS
    {
      u16* gm = const_cast<u16*>(ag + k0);
      u16* lm = &As[tid][0];
      __builtin_amdgcn_global_load_async_to_lds_b128((AS1 v4i*)(gm + 0),  (AS3 v4i*)(lm + 0),  0, 0);
      __builtin_amdgcn_global_load_async_to_lds_b128((AS1 v4i*)(gm + 8),  (AS3 v4i*)(lm + 8),  0, 0);
      __builtin_amdgcn_global_load_async_to_lds_b128((AS1 v4i*)(gm + 16), (AS3 v4i*)(lm + 16), 0, 0);
      __builtin_amdgcn_global_load_async_to_lds_b128((AS1 v4i*)(gm + 24), (AS3 v4i*)(lm + 24), 0, 0);
    }
#else
    {
      const u16* g = ag + k0;
      *(uint4*)&As[tid][0]  = *(const uint4*)g;
      *(uint4*)&As[tid][8]  = *(const uint4*)(g + 8);
      *(uint4*)&As[tid][16] = *(const uint4*)(g + 16);
      *(uint4*)&As[tid][24] = *(const uint4*)(g + 24);
    }
#endif
    // ---- stage B transposed via 4x4 register transpose ----
    {
      const u16* g = Bw + (long)(k0 + kb) * N + nbase + cb;
      u64 r0 = *(const u64*)g;
      u64 r1 = *(const u64*)(g + N);
      u64 r2 = *(const u64*)(g + 2 * N);
      u64 r3 = *(const u64*)(g + 3 * N);
      #pragma unroll
      for (int c = 0; c < 4; ++c)
        *(u64*)&Bs[cb + c][kb] = tr4(r0, r1, r2, r3, c);
    }
    // prefetch next K-tile
    if (k0 + 32 < K) {
      __builtin_prefetch(ag + k0 + 32, 0, 0);
      __builtin_prefetch(Bw + (long)(k0 + 32 + kb) * N + nbase + cb, 0, 0);
    }
#if HAVE_ASYNC_LDS
    asm volatile("s_wait_asynccnt 0x0" ::: "memory");
#endif
    __syncthreads();

    // ---- compute: wave covers rows [32*wave, 32*wave+32) ----
    const int rb = wave * 32;
    v16bf af0 = frag16(&As[rb + ln][hf * 8],      &As[rb + ln][16 + hf * 8]);
    v16bf af1 = frag16(&As[rb + 16 + ln][hf * 8], &As[rb + 16 + ln][16 + hf * 8]);
    #pragma unroll
    for (int g = 0; g < 4; ++g) {
      v16bf bf = frag16(&Bs[g * 16 + ln][hf * 8], &Bs[g * 16 + ln][16 + hf * 8]);
      acc[0][g] = __builtin_amdgcn_wmma_f32_16x16x32_bf16(
          false, af0, false, bf, (short)0, acc[0][g], false, false);
      acc[1][g] = __builtin_amdgcn_wmma_f32_16x16x32_bf16(
          false, af1, false, bf, (short)0, acc[1][g], false, false);
    }
    __syncthreads();
  }

  // C layout: VGPR r -> M = r + 8*hf, N = ln
  #pragma unroll
  for (int st = 0; st < 2; ++st) {
    #pragma unroll
    for (int g = 0; g < 4; ++g) {
      #pragma unroll
      for (int r = 0; r < 8; ++r) {
        long m = mbase + wave * 32 + st * 16 + r + 8 * hf;
        long n = nbase + g * 16 + ln;
        float v = acc[st][g][r];
        if (bias) v += bias[n];
        if (relu) v = v > 0.f ? v : 0.f;
        long orow = (Tdst > 0) ? (m / Tsrc) * (long)Tdst + (m % Tsrc) : m;
        if (outF) outF[orow * N + n] = v;
        if (outB) outB[orow * N + n] = f2bf(v);
      }
    }
  }
}

// ---------------- fused attention (one wave per b,h,16-query tile) ----------------
// Q/K/V in per-batch padded (TP rows) bf16 buffers; output compact (b*Tq+q) bf16.
__global__ __launch_bounds__(32)
void attn_kernel(const u16* __restrict__ Qp, const u16* __restrict__ Kp,
                 const u16* __restrict__ Vp, u16* __restrict__ Out,
                 const int* __restrict__ mask, int Tq, int Tk, int causal)
{
  __shared__ __align__(16) float Sf[16][TP];   // raw scores
  __shared__ __align__(16) u16   Pb[16][TP];   // exp(S - max) as bf16
  __shared__ __align__(16) u16   Vt[64][48];   // V^T tile: [head-dim][32 keys]
  __shared__ float rinv[16];

  const int lane = threadIdx.x;
  const int hf = lane >> 4, ln = lane & 15;
  const int qt = blockIdx.x, h = blockIdx.y, b = blockIdx.z;
  const int qbase = qt * 16;
  const float scale = 0.125f;                  // HD^-0.5

  // Q fragments for the two K=32 steps over head dim (loaded once)
  v16bf qf[2];
  #pragma unroll
  for (int s = 0; s < 2; ++s) {
    const u16* p = Qp + ((long)b * TP + qbase + ln) * NE + (long)h * HD + 32 * s;
    qf[s] = frag16(p + hf * 8, p + 16 + hf * 8);
  }

  // ---- S = scale * Q K^T, masked, into LDS ----
  for (int kt = 0; kt < TP / 16; ++kt) {
    const int kbq = kt * 16;
    v8f sa = {};
    #pragma unroll
    for (int s = 0; s < 2; ++s) {
      // B = K^T: column n of B is K-row (kbq+n), contiguous in global memory
      const u16* p = Kp + ((long)b * TP + kbq + ln) * NE + (long)h * HD + 32 * s;
      v16bf kf = frag16(p + hf * 8, p + 16 + hf * 8);
      sa = __builtin_amdgcn_wmma_f32_16x16x32_bf16(
          false, qf[s], false, kf, (short)0, sa, false, false);
    }
    const int k = kbq + ln;
    const bool kval = (k < Tk) && (mask[(long)b * Tk + (k < Tk ? k : 0)] != 0);
    #pragma unroll
    for (int r = 0; r < 8; ++r) {
      const int q = qbase + r + 8 * hf;
      float v = sa[r] * scale;
      if (!kval || (causal && k > q)) v = NEGV;
      Sf[r + 8 * hf][k] = v;
    }
  }
  __syncthreads();

  // ---- softmax per row (wave-wide shfl reduction) ----
  for (int r = 0; r < 16; ++r) {
    float mx = NEGV;
    #pragma unroll
    for (int i = 0; i < TP / 32; ++i) mx = fmaxf(mx, Sf[r][lane + 32 * i]);
    #pragma unroll
    for (int o = 16; o > 0; o >>= 1) mx = fmaxf(mx, __shfl_xor(mx, o, 32));
    float sm = 0.f;
    #pragma unroll
    for (int i = 0; i < TP / 32; ++i) {
      float p = __expf(Sf[r][lane + 32 * i] - mx);
      sm += p;
      Pb[r][lane + 32 * i] = f2bf(p);
    }
    #pragma unroll
    for (int o = 16; o > 0; o >>= 1) sm += __shfl_xor(sm, o, 32);
    if (lane == 0) rinv[r] = 1.0f / sm;
  }
  __syncthreads();

  // ---- O = P V ----
  v8f oa[4] = { v8f{}, v8f{}, v8f{}, v8f{} };
  for (int s = 0; s < TP / 32; ++s) {
    // stage V^T via 4x4 register transposes (pad keys zeroed: no garbage*0 NaN)
    #pragma unroll
    for (int j = 0; j < 4; ++j) {
      const int bi = lane + 32 * j;        // 0..127 blocks of 4keys x 4dims
      const int kk = (bi & 7) * 4;
      const int db = (bi >> 3) * 4;
      u64 r[4];
      #pragma unroll
      for (int t = 0; t < 4; ++t) {
        const int kr = 32 * s + kk + t;
        r[t] = (kr < Tk)
          ? *(const u64*)(Vp + ((long)b * TP + kr) * NE + (long)h * HD + db)
          : 0ull;
      }
      #pragma unroll
      for (int c = 0; c < 4; ++c)
        *(u64*)&Vt[db + c][kk] = tr4(r[0], r[1], r[2], r[3], c);
    }
    __syncthreads();
    const u16* pp = &Pb[ln][32 * s + hf * 8];
    v16bf pf = frag16(pp, pp + 16);
    #pragma unroll
    for (int g = 0; g < 4; ++g) {
      const u16* bp = &Vt[g * 16 + ln][hf * 8];
      v16bf vf = frag16(bp, bp + 16);
      oa[g] = __builtin_amdgcn_wmma_f32_16x16x32_bf16(
          false, pf, false, vf, (short)0, oa[g], false, false);
    }
    __syncthreads();
  }

  // ---- store (compact rows), normalize by row sum ----
  #pragma unroll
  for (int g = 0; g < 4; ++g) {
    #pragma unroll
    for (int r = 0; r < 8; ++r) {
      const int q = qbase + r + 8 * hf;
      if (q < Tq) {
        const float v = oa[g][r] * rinv[r + 8 * hf];
        Out[((long)b * Tq + q) * NE + (long)h * HD + g * 16 + ln] = f2bf(v);
      }
    }
  }
}

// ---------------- fused residual add + LayerNorm ----------------
__global__ __launch_bounds__(256)
void add_ln_kernel(const float* __restrict__ P, const float* __restrict__ R,
                   const float* __restrict__ g, const float* __restrict__ be,
                   float* __restrict__ outF, u16* __restrict__ outB)
{
  __shared__ float red[8];
  const long row = blockIdx.x;
  const int tid = threadIdx.x;
  const float v0 = P[row * NE + tid]       + R[row * NE + tid];
  const float v1 = P[row * NE + tid + 256] + R[row * NE + tid + 256];

  float s = v0 + v1;
  #pragma unroll
  for (int o = 16; o > 0; o >>= 1) s += __shfl_xor(s, o, 32);
  if ((tid & 31) == 0) red[tid >> 5] = s;
  __syncthreads();
  float tot = 0.f;
  #pragma unroll
  for (int i = 0; i < 8; ++i) tot += red[i];
  const float mean = tot * (1.0f / NE);

  const float d0 = v0 - mean, d1 = v1 - mean;
  s = d0 * d0 + d1 * d1;
  #pragma unroll
  for (int o = 16; o > 0; o >>= 1) s += __shfl_xor(s, o, 32);
  __syncthreads();
  if ((tid & 31) == 0) red[tid >> 5] = s;
  __syncthreads();
  tot = 0.f;
  #pragma unroll
  for (int i = 0; i < 8; ++i) tot += red[i];
  const float inv = rsqrtf(tot * (1.0f / NE) + 1e-5f);

  const float o0 = d0 * inv * g[tid]       + be[tid];
  const float o1 = d1 * inv * g[tid + 256] + be[tid + 256];
  if (outF) { outF[row * NE + tid] = o0; outF[row * NE + tid + 256] = o1; }
  if (outB) { outB[row * NE + tid] = f2bf(o0); outB[row * NE + tid + 256] = f2bf(o1); }
}

// ---------------- host orchestration ----------------
extern "C" void kernel_launch(void* const* d_in, const int* in_sizes, int n_in,
                              void* d_out, int out_size, void* d_ws, size_t ws_size,
                              hipStream_t stream) {
  const float* x        = (const float*)d_in[0];
  const float* enc      = (const float*)d_in[1];
  const int*   min_mask = (const int*)d_in[2];
  const int*   mout     = (const int*)d_in[3];
  const float* Wq_sa = (const float*)d_in[4];
  const float* Wk_sa = (const float*)d_in[5];
  const float* Wv_sa = (const float*)d_in[6];
  const float* Wo_sa = (const float*)d_in[7];
  const float* bo_sa = (const float*)d_in[8];
  const float* Wq_ca = (const float*)d_in[9];
  const float* Wk_ca = (const float*)d_in[10];
  const float* Wv_ca = (const float*)d_in[11];
  const float* Wo_ca = (const float*)d_in[12];
  const float* bo_ca = (const float*)d_in[13];
  const float* W1 = (const float*)d_in[14];
  const float* b1 = (const float*)d_in[15];
  const float* W2 = (const float*)d_in[16];
  const float* b2 = (const float*)d_in[17];
  const float* g1  = (const float*)d_in[18];
  const float* be1 = (const float*)d_in[19];
  const float* gc  = (const float*)d_in[20];
  const float* bec = (const float*)d_in[21];
  const float* g2  = (const float*)d_in[22];
  const float* be2 = (const float*)d_in[23];
  float* out = (float*)d_out;

  char* ws = (char*)d_ws;
  size_t off = 0;
  auto carve = [&](size_t bytes) -> void* {
    void* p = ws + off;
    off = (off + bytes + 255) & ~(size_t)255;
    return p;
  };
  const size_t ACT_B = (size_t)MTOK * NE;           // elems (compact activations)
  const size_t PAD_B = (size_t)BB * TP * NE;        // elems (padded per batch)

  u16* wqsa = (u16*)carve(NE * NE * 2);
  u16* wksa = (u16*)carve(NE * NE * 2);
  u16* wvsa = (u16*)carve(NE * NE * 2);
  u16* wosa = (u16*)carve(NE * NE * 2);
  u16* wqca = (u16*)carve(NE * NE * 2);
  u16* wkca = (u16*)carve(NE * NE * 2);
  u16* wvca = (u16*)carve(NE * NE * 2);
  u16* woca = (u16*)carve(NE * NE * 2);
  u16* w1b  = (u16*)carve((size_t)NE * 4 * NE * 2);
  u16* w2b  = (u16*)carve((size_t)4 * NE * NE * 2);
  u16* Xb   = (u16*)carve(ACT_B * 2);
  u16* Eb   = (u16*)carve(ACT_B * 2);
  u16* Attb = (u16*)carve(ACT_B * 2);
  u16* Y1b  = (u16*)carve(ACT_B * 2);
  u16* Y2b  = (u16*)carve(ACT_B * 2);
  u16* Qp   = (u16*)carve(PAD_B * 2);
  u16* Kp   = (u16*)carve(PAD_B * 2);
  u16* Vp   = (u16*)carve(PAD_B * 2);
  u16* Hb   = (u16*)carve((size_t)MTOK * 4 * NE * 2);
  float* Pf  = (float*)carve(ACT_B * 4);
  float* Y1f = (float*)carve(ACT_B * 4);
  float* Y2f = (float*)carve(ACT_B * 4);

  auto cast = [&](const float* s, u16* d, long n) {
    cast_f32_bf16<<<dim3(1024), dim3(256), 0, stream>>>(s, d, (int)n);
  };
  cast(x,   Xb, (long)ACT_B);
  cast(enc, Eb, (long)ACT_B);
  cast(Wq_sa, wqsa, NE * NE); cast(Wk_sa, wksa, NE * NE);
  cast(Wv_sa, wvsa, NE * NE); cast(Wo_sa, wosa, NE * NE);
  cast(Wq_ca, wqca, NE * NE); cast(Wk_ca, wkca, NE * NE);
  cast(Wv_ca, wvca, NE * NE); cast(Wo_ca, woca, NE * NE);
  cast(W1, w1b, (long)NE * 4 * NE);
  cast(W2, w2b, (long)4 * NE * NE);

  auto gemm = [&](const u16* A, const u16* Bw, const float* bias,
                  float* oF, u16* oB, int N, int K, int relu, int remap) {
    gemm_bf16<<<dim3(MTOK / 128, N / 64), dim3(128), 0, stream>>>(
        A, Bw, bias, oF, oB, MTOK, N, K, relu, TT, remap ? TP : 0);
  };

  // ---- self attention ----
  gemm(Xb, wqsa, nullptr, nullptr, Qp, NE, NE, 0, 1);
  gemm(Xb, wksa, nullptr, nullptr, Kp, NE, NE, 0, 1);
  gemm(Xb, wvsa, nullptr, nullptr, Vp, NE, NE, 0, 1);
  attn_kernel<<<dim3(TP / 16, NH, BB), dim3(32), 0, stream>>>(
      Qp, Kp, Vp, Attb, mout, TT, TT, /*causal=*/1);
  gemm(Attb, wosa, bo_sa, Pf, nullptr, NE, NE, 0, 0);
  add_ln_kernel<<<dim3(MTOK), dim3(256), 0, stream>>>(Pf, x, g1, be1, Y1f, Y1b);

  // ---- cross attention ----
  gemm(Y1b, wqca, nullptr, nullptr, Qp, NE, NE, 0, 1);
  gemm(Eb,  wkca, nullptr, nullptr, Kp, NE, NE, 0, 1);
  gemm(Eb,  wvca, nullptr, nullptr, Vp, NE, NE, 0, 1);
  attn_kernel<<<dim3(TP / 16, NH, BB), dim3(32), 0, stream>>>(
      Qp, Kp, Vp, Attb, min_mask, TT, TT, /*causal=*/0);
  gemm(Attb, woca, bo_ca, Pf, nullptr, NE, NE, 0, 0);
  add_ln_kernel<<<dim3(MTOK), dim3(256), 0, stream>>>(Pf, Y1f, gc, bec, Y2f, Y2b);

  // ---- FFN ----
  gemm(Y2b, w1b, b1, nullptr, Hb, 4 * NE, NE, /*relu=*/1, 0);
  gemm(Hb,  w2b, b2, Pf, nullptr, NE, 4 * NE, 0, 0);
  add_ln_kernel<<<dim3(MTOK), dim3(256), 0, stream>>>(Pf, Y2f, g2, be2, out, nullptr);
}
